// HeteroSAGE_16767552323881
// MI455X (gfx1250) — compile-verified, matrix operand order checked by reading, not run on hardware
//
#include <hip/hip_runtime.h>
#include <hip/hip_bf16.h>

// ---------------------------------------------------------------------------
// Types for CDNA5 WMMA (wave32, 16x16x32 bf16 -> f32)
// ---------------------------------------------------------------------------
typedef __attribute__((ext_vector_type(16))) __bf16 v16bf;
typedef __attribute__((ext_vector_type(8)))  float  v8f;

union Frag16 {
    v16bf        v;
    unsigned int u[8];
};

__device__ __forceinline__ unsigned int pack2_bf16(float a, float b) {
    unsigned int ua = __float_as_uint(a);
    unsigned int ub = __float_as_uint(b);
    ua += 0x7FFFu + ((ua >> 16) & 1u);   // round-to-nearest-even
    ub += 0x7FFFu + ((ub >> 16) & 1u);
    return (ua >> 16) | (ub & 0xFFFF0000u);
}

// ---------------------------------------------------------------------------
// Preswizzle a weight matrix B[K,N-slice] (f32, row-major, ld=ldb) into the
// WMMA bf16 fragment layout, zero-padded beyond K:
//   out[((kt*16)+cp)*Ntot + colOff + n] = pack(B[k][n], B[k+1][n])
//   with k = kt*32 + (cp&3)*2 + ((cp>>2)&1)*8 + ((cp>>3)&1)*16
// A lane reading vgpr v with half 'hi' uses cp = (v&3) + hi*4 + (v>>2)*8.
// ---------------------------------------------------------------------------
__global__ __launch_bounds__(256)
void preswizzle_b(const float* __restrict__ B, int ldb, int K, int KT, int Ncols,
                  unsigned int* __restrict__ out, int Ntot, int colOff)
{
    long long idx   = (long long)blockIdx.x * 256 + threadIdx.x;
    long long total = (long long)KT * 16 * Ncols;
    if (idx >= total) return;
    int n  = (int)(idx % Ncols);
    int cp = (int)((idx / Ncols) & 15);
    int kt = (int)(idx / ((long long)Ncols * 16));
    int k  = kt * 32 + (cp & 3) * 2 + ((cp >> 2) & 1) * 8 + ((cp >> 3) & 1) * 16;
    float b0 = (k     < K) ? B[(long long)k * ldb + n]       : 0.f;
    float b1 = (k + 1 < K) ? B[(long long)(k + 1) * ldb + n] : 0.f;
    out[((long long)kt * 16 + cp) * Ntot + colOff + n] = pack2_bf16(b0, b1);
}

// ---------------------------------------------------------------------------
// GEMM: C[M,N] (+)= A[M,K] @ B[K,N], A f32 row-major, B preswizzled bf16.
// Block = 256 threads (8 waves), tile 64(M) x 128(N), BK = 32.
// Wave grid 2x4; each wave owns 32x32 (2x2 WMMA tiles).
// REQUIRES: N % 128 == 0 (true for all call sites: 128/256/512).
// grid = (N/128, ceil(M/64))
// ---------------------------------------------------------------------------
__global__ __launch_bounds__(256)
void gemm_bf16_wmma(const float* __restrict__ A, int lda,
                    const unsigned int* __restrict__ Bpre,
                    float* __restrict__ C, int ldc,
                    int M, int N, int K, int accum)
{
    __shared__ unsigned int sA[64][16];   // 64 rows x 16 packed bf16 k-pairs

    const int tid  = threadIdx.x;
    const int lane = tid & 31;
    const int wave = tid >> 5;
    const int wr   = wave >> 2;          // 0..1 -> 32 rows each
    const int wc   = wave & 3;           // 0..3 -> 32 cols each
    const int lm   = lane & 15;
    const int hi   = lane >> 4;

    const int rowBase = blockIdx.y * 64;
    const int colBase = blockIdx.x * 128;
    const int KT      = (K + 31) >> 5;
    const bool rowsFull = (rowBase + 64 <= M);

    v8f acc[2][2];
    #pragma unroll
    for (int i = 0; i < 2; ++i)
        #pragma unroll
        for (int j = 0; j < 2; ++j)
            acc[i][j] = (v8f){0.f, 0.f, 0.f, 0.f, 0.f, 0.f, 0.f, 0.f};

    for (int kt = 0; kt < KT; ++kt) {
        const int k0 = kt << 5;
        if (rowsFull && (k0 + 32 <= K)) {
            // branchless fast path: 4 float2 loads per thread
            #pragma unroll
            for (int i = 0; i < 4; ++i) {
                int idx = i * 256 + tid;
                int r   = idx >> 4;
                int cp  = idx & 15;
                const float2 f =
                    *(const float2*)(A + (long long)(rowBase + r) * lda + k0 + cp * 2);
                sA[r][cp] = pack2_bf16(f.x, f.y);
            }
            if (k0 + 32 < K) {  // prefetch next A tile (one line per thread)
                int r  = tid >> 4;
                int cp = tid & 15;
                __builtin_prefetch(
                    A + (long long)(rowBase + r) * lda + k0 + 32 + cp * 2, 0, 1);
            }
        } else {
            #pragma unroll
            for (int i = 0; i < 4; ++i) {
                int idx = i * 256 + tid;
                int r   = idx >> 4;
                int cp  = idx & 15;
                int gr  = rowBase + r;
                int k   = k0 + cp * 2;
                float a0 = 0.f, a1 = 0.f;
                if (gr < M) {
                    const float* ap = A + (long long)gr * lda;
                    if (k     < K) a0 = ap[k];
                    if (k + 1 < K) a1 = ap[k + 1];
                }
                sA[r][cp] = pack2_bf16(a0, a1);
            }
        }
        __syncthreads();

        const unsigned int* bb = Bpre + (long long)kt * 16 * N;
        Frag16 aF[2], bF[2];
        #pragma unroll
        for (int t = 0; t < 2; ++t) {
            const int m = wr * 32 + t * 16 + lm;
            const int n = colBase + wc * 32 + t * 16 + lm;
            #pragma unroll
            for (int v = 0; v < 8; ++v) {
                const int cp = (v & 3) + (hi << 2) + ((v >> 2) << 3);
                aF[t].u[v] = sA[m][cp];
                bF[t].u[v] = bb[(long long)cp * N + n];
            }
        }
        #pragma unroll
        for (int i = 0; i < 2; ++i)
            #pragma unroll
            for (int j = 0; j < 2; ++j)
                acc[i][j] = __builtin_amdgcn_wmma_f32_16x16x32_bf16(
                    false, aF[i].v, false, bF[j].v,
                    (short)0, acc[i][j], false, false);
        __syncthreads();
    }

    // store: vgpr v, lane l -> m = v + 8*hi, n = lm  (N always fully covered)
    if (accum) {
        #pragma unroll
        for (int i = 0; i < 2; ++i)
            #pragma unroll
            for (int j = 0; j < 2; ++j)
                #pragma unroll
                for (int v = 0; v < 8; ++v) {
                    int m = rowBase + wr * 32 + i * 16 + v + hi * 8;
                    if (m < M) {
                        long long off =
                            (long long)m * ldc + colBase + wc * 32 + j * 16 + lm;
                        C[off] += acc[i][j][v];
                    }
                }
    } else {
        #pragma unroll
        for (int i = 0; i < 2; ++i)
            #pragma unroll
            for (int j = 0; j < 2; ++j)
                #pragma unroll
                for (int v = 0; v < 8; ++v) {
                    int m = rowBase + wr * 32 + i * 16 + v + hi * 8;
                    if (m < M) {
                        long long off =
                            (long long)m * ldc + colBase + wc * 32 + j * 16 + lm;
                        C[off] = acc[i][j][v];
                    }
                }
    }
}

// ---------------------------------------------------------------------------
// Attention combine. XW layout: [row = n*5+s][512] with col = h*128 + o.
// One block (256 threads) per node.
// ---------------------------------------------------------------------------
__global__ __launch_bounds__(256)
void attn_combine(const float* __restrict__ XW, const float* __restrict__ q_att,
                  float* __restrict__ XS)
{
    __shared__ float logit[20];
    __shared__ float wsm[20];
    const int n    = blockIdx.x;
    const int tid  = threadIdx.x;
    const int lane = tid & 31;
    const int wave = tid >> 5;

    for (int p = wave; p < 20; p += 8) {
        int h = p / 5, s = p % 5;
        const float* row = XW + (long long)(n * 5 + s) * 512 + h * 128;
        const float* q   = q_att + h * 128;
        float partial = 0.f;
        #pragma unroll
        for (int it = 0; it < 4; ++it) {
            int d = lane + it * 32;
            partial += row[d] * q[d];
        }
        #pragma unroll
        for (int off = 16; off > 0; off >>= 1)
            partial += __shfl_xor(partial, off, 32);
        if (lane == 0)
            logit[p] = partial > 0.f ? partial : 0.2f * partial;
    }
    __syncthreads();
    if (tid < 4) {
        int h = tid;
        float mx = -1e30f;
        #pragma unroll
        for (int s = 0; s < 5; ++s) mx = fmaxf(mx, logit[h * 5 + s]);
        float e[5], sum = 0.f;
        #pragma unroll
        for (int s = 0; s < 5; ++s) { e[s] = __expf(logit[h * 5 + s] - mx); sum += e[s]; }
        #pragma unroll
        for (int s = 0; s < 5; ++s) wsm[h * 5 + s] = e[s] / sum;
    }
    __syncthreads();
    #pragma unroll
    for (int rep = 0; rep < 2; ++rep) {
        int od = tid + rep * 256;
        int h  = od >> 7;
        const float* base = XW + (long long)n * 5 * 512 + od;
        float a = 0.f;
        #pragma unroll
        for (int s = 0; s < 5; ++s)
            a += wsm[h * 5 + s] * base[s * 512];
        XS[(long long)n * 512 + od] = a;
    }
}

// ---------------------------------------------------------------------------
// Edge scatter (segment-sum): one block per edge, 256 dims per message
// ---------------------------------------------------------------------------
__global__ __launch_bounds__(256)
void edge_scatter(const float* __restrict__ msg, const int* __restrict__ src,
                  const int* __restrict__ tgt, float* __restrict__ acc,
                  float* __restrict__ cnt)
{
    int e = blockIdx.x;
    int d = threadIdx.x;
    int s = src[e], t = tgt[e];
    atomicAdd(&acc[(long long)t * 256 + d], msg[(long long)s * 256 + d]);
    if (d == 0) atomicAdd(&cnt[t], 1.0f);
}

__global__ void fill_zero(float* __restrict__ p, long long n)
{
    long long i      = (long long)blockIdx.x * blockDim.x + threadIdx.x;
    long long stride = (long long)gridDim.x * blockDim.x;
    for (; i < n; i += stride) p[i] = 0.f;
}

// out = relu(scale * (T + acc1/max(cnt1,1) [+ acc2/max(cnt2,1)]))
__global__ __launch_bounds__(256)
void finalize_node(const float* __restrict__ T,
                   const float* __restrict__ acc1, const float* __restrict__ cnt1,
                   const float* __restrict__ acc2, const float* __restrict__ cnt2,
                   float scale, float* __restrict__ out)
{
    long long i = (long long)blockIdx.x * 256 + threadIdx.x;
    int n = (int)(i >> 8);
    float v = T[i] + acc1[i] / fmaxf(cnt1[n], 1.0f);
    if (acc2) v += acc2[i] / fmaxf(cnt2[n], 1.0f);
    v *= scale;
    out[i] = v > 0.f ? v : 0.f;
}

__global__ __launch_bounds__(256)
void avg_kernel(const float* __restrict__ a, const float* __restrict__ b,
                float* __restrict__ out, long long n)
{
    long long i = (long long)blockIdx.x * 256 + threadIdx.x;
    if (i < n) out[i] = 0.5f * (a[i] + b[i]);
}

// warp-per-node dot(X[n,:256], w) + b
__global__ __launch_bounds__(256)
void linear_out(const float* __restrict__ X, const float* __restrict__ w,
                const float* __restrict__ b, float* __restrict__ out, int nNodes)
{
    int node = blockIdx.x * 8 + (threadIdx.x >> 5);
    int lane = threadIdx.x & 31;
    if (node >= nNodes) return;
    const float* row = X + (long long)node * 256;
    float p = 0.f;
    #pragma unroll
    for (int it = 0; it < 8; ++it) {
        int d = lane + it * 32;
        p += row[d] * w[d];
    }
    #pragma unroll
    for (int off = 16; off > 0; off >>= 1)
        p += __shfl_xor(p, off, 32);
    if (lane == 0) out[node] = p + b[0];
}

// ---------------------------------------------------------------------------
// Host-side orchestration
// ---------------------------------------------------------------------------
static const int N_SPOT = 20000, N_CITY = 2000, N_WORD = 20000;
static const int E_HAS = 100000, E_IN = 100000, E_DESC = 400000, E_RDESC = 400000;
static const int HID = 256;

static inline void launch_gemm(const float* A, int lda, const unsigned int* Bpre,
                               float* C, int ldc, int M, int N, int K, int accum,
                               hipStream_t st)
{
    dim3 grid(N / 128, (M + 63) / 64);
    gemm_bf16_wmma<<<grid, 256, 0, st>>>(A, lda, Bpre, C, ldc, M, N, K, accum);
}

static inline void launch_preswz(const float* B, int ldb, int K, int Ncols,
                                 unsigned int* out, int Ntot, int colOff,
                                 hipStream_t st)
{
    int KT = (K + 31) / 32;
    long long total = (long long)KT * 16 * Ncols;
    preswizzle_b<<<(unsigned)((total + 255) / 256), 256, 0, st>>>(
        B, ldb, K, KT, Ncols, out, Ntot, colOff);
}

// One hetero-conv layer.
// wp[8]/Kw_[8] = preswizzled {has_s, has_t, in_s, in_t, desc_s, desc_t, rdesc_s, rdesc_t}
static void run_conv_layer(const float* Xs, int Ks, const float* Xc, int Kc,
                           const float* Xw, int Kw, unsigned int* const* wp,
                           const int* eHs, const int* eHt, const int* eIs, const int* eIt,
                           const int* eDs, const int* eDt, const int* eRs, const int* eRt,
                           float* scratch, float* Sout, float* Cout, float* Wout,
                           hipStream_t st)
{
    const long long SPOTF = (long long)N_SPOT * HID;
    const long long CITYF = (long long)N_CITY * HID;
    const long long WORDF = (long long)N_WORD * HID;

    float* p       = scratch;
    float* Tspot   = p; p += SPOTF;
    float* Tcity   = p; p += CITYF;
    float* Tword   = p; p += WORDF;
    float* Mhas    = p; p += CITYF;
    float* Mdesc   = p; p += WORDF;
    float* Min     = p; p += SPOTF;
    float* Mrdesc  = p; p += SPOTF;
    float* AccHas  = p; p += SPOTF;
    float* AccDesc = p; p += SPOTF;
    float* AccIn   = p; p += CITYF;
    float* AccRd   = p; p += WORDF;
    float* CntHas  = p; p += N_SPOT;
    float* CntDesc = p; p += N_SPOT;
    float* CntIn   = p; p += N_CITY;
    float* CntRd   = p; p += N_WORD;

    fill_zero<<<2048, 256, 0, st>>>(AccHas, (long long)(p - AccHas));

    // target-side transforms
    launch_gemm(Xs, Ks, wp[1], Tspot, HID, N_SPOT, HID, Ks, 0, st);
    launch_gemm(Xs, Ks, wp[5], Tspot, HID, N_SPOT, HID, Ks, 1, st);
    launch_gemm(Xc, Kc, wp[3], Tcity, HID, N_CITY, HID, Kc, 0, st);
    launch_gemm(Xw, Kw, wp[7], Tword, HID, N_WORD, HID, Kw, 0, st);
    // source-side messages
    launch_gemm(Xc, Kc, wp[0], Mhas,   HID, N_CITY, HID, Kc, 0, st);
    launch_gemm(Xw, Kw, wp[4], Mdesc,  HID, N_WORD, HID, Kw, 0, st);
    launch_gemm(Xs, Ks, wp[2], Min,    HID, N_SPOT, HID, Ks, 0, st);
    launch_gemm(Xs, Ks, wp[6], Mrdesc, HID, N_SPOT, HID, Ks, 0, st);

    edge_scatter<<<E_HAS,   256, 0, st>>>(Mhas,   eHs, eHt, AccHas,  CntHas);
    edge_scatter<<<E_DESC,  256, 0, st>>>(Mdesc,  eDs, eDt, AccDesc, CntDesc);
    edge_scatter<<<E_IN,    256, 0, st>>>(Min,    eIs, eIt, AccIn,   CntIn);
    edge_scatter<<<E_RDESC, 256, 0, st>>>(Mrdesc, eRs, eRt, AccRd,   CntRd);

    finalize_node<<<N_SPOT, 256, 0, st>>>(Tspot, AccHas, CntHas, AccDesc, CntDesc, 0.5f, Sout);
    finalize_node<<<N_CITY, 256, 0, st>>>(Tcity, AccIn,  CntIn,  nullptr, nullptr, 1.0f, Cout);
    finalize_node<<<N_WORD, 256, 0, st>>>(Tword, AccRd,  CntRd,  nullptr, nullptr, 1.0f, Wout);
}

extern "C" void kernel_launch(void* const* d_in, const int* in_sizes, int n_in,
                              void* d_out, int out_size, void* d_ws, size_t ws_size,
                              hipStream_t stream)
{
    const float* x_spot = (const float*)d_in[0];
    const float* x_city = (const float*)d_in[1];
    const float* x_word = (const float*)d_in[2];
    const float* W_att  = (const float*)d_in[3];
    const float* q_att  = (const float*)d_in[4];
    const float* w1[8], *w2[8];
    for (int i = 0; i < 8; ++i) { w1[i] = (const float*)d_in[5 + i]; w2[i] = (const float*)d_in[13 + i]; }
    const float* wl_spot = (const float*)d_in[21];
    const float* wl_city = (const float*)d_in[22];
    const float* wl_word = (const float*)d_in[23];
    const float* bl_spot = (const float*)d_in[24];
    const float* bl_city = (const float*)d_in[25];
    const float* bl_word = (const float*)d_in[26];
    const int* eHs = (const int*)d_in[27]; const int* eHt = (const int*)d_in[28];
    const int* eIs = (const int*)d_in[29]; const int* eIt = (const int*)d_in[30];
    const int* eDs = (const int*)d_in[31]; const int* eDt = (const int*)d_in[32];
    const int* eRs = (const int*)d_in[33]; const int* eRt = (const int*)d_in[34];

    float* ws = (float*)d_ws;
    const long long SPOTF = (long long)N_SPOT * HID;
    const long long CITYF = (long long)N_CITY * HID;
    const long long WORDF = (long long)N_WORD * HID;

    long long o = 0;
    float* XS  = ws + o; o += (long long)N_SPOT * 512;   // attention output
    float* XW  = ws + o; o += 100000LL * 512;            // head projections / layer scratch
    float* S1  = ws + o; o += SPOTF;
    float* C1  = ws + o; o += CITYF;
    float* W1b = ws + o; o += WORDF;
    float* S2  = ws + o; o += SPOTF;
    float* C2  = ws + o; o += CITYF;
    float* W2b = ws + o; o += WORDF;

    // preswizzled bf16 weights (unsigned int == 4B, carve from float ws)
    const int K1[8] = {128, 512, 512, 128, 300, 512, 512, 300};
    unsigned int* BpreAtt = (unsigned int*)(ws + o);
    o += 16LL * 16 * 512;                                // KT=16, Ntot=512
    unsigned int* w1p[8];
    for (int i = 0; i < 8; ++i) {
        w1p[i] = (unsigned int*)(ws + o);
        o += (long long)((K1[i] + 31) / 32) * 16 * HID;
    }
    unsigned int* w2p[8];
    for (int i = 0; i < 8; ++i) {
        w2p[i] = (unsigned int*)(ws + o);
        o += (long long)((HID + 31) / 32) * 16 * HID;
    }

    // ---- preswizzle all weights ------------------------------------------
    for (int h = 0; h < 4; ++h)
        launch_preswz(W_att + (long long)h * 512 * 128, 128, 512, 128,
                      BpreAtt, 512, h * 128, stream);
    for (int i = 0; i < 8; ++i) {
        launch_preswz(w1[i], HID, K1[i], HID, w1p[i], HID, 0, stream);
        launch_preswz(w2[i], HID, HID,   HID, w2p[i], HID, 0, stream);
    }

    // ---- attention: (100000 x 512) @ (512 x 512 fused heads) -------------
    launch_gemm(x_spot, 512, BpreAtt, XW, 512, 100000, 512, 512, 0, stream);
    attn_combine<<<N_SPOT, 256, 0, stream>>>(XW, q_att, XS);

    // ---- layer 1 (XW region is dead now; reuse as scratch) ---------------
    run_conv_layer(XS, 512, x_city, 128, x_word, 300, w1p,
                   eHs, eHt, eIs, eIt, eDs, eDt, eRs, eRt,
                   XW, S1, C1, W1b, stream);
    // ---- layer 2
    run_conv_layer(S1, 256, C1, 256, W1b, 256, w2p,
                   eHs, eHt, eIs, eIt, eDs, eDt, eRs, eRt,
                   XW, S2, C2, W2b, stream);

    // ---- outputs ---------------------------------------------------------
    float* out = (float*)d_out;
    float* oS  = out;
    float* oC  = oS + SPOTF;
    float* oW  = oC + CITYF;
    float* oSl = oW + WORDF;
    float* oCl = oSl + N_SPOT;
    float* oWl = oCl + N_CITY;

    avg_kernel<<<(unsigned)((SPOTF + 255) / 256), 256, 0, stream>>>(S1, S2, oS, SPOTF);
    avg_kernel<<<(unsigned)((CITYF + 255) / 256), 256, 0, stream>>>(C1, C2, oC, CITYF);
    avg_kernel<<<(unsigned)((WORDF + 255) / 256), 256, 0, stream>>>(W1b, W2b, oW, WORDF);

    linear_out<<<(N_SPOT + 7) / 8, 256, 0, stream>>>(S2,  wl_spot, bl_spot, oSl, N_SPOT);
    linear_out<<<(N_CITY + 7) / 8, 256, 0, stream>>>(C2,  wl_city, bl_city, oCl, N_CITY);
    linear_out<<<(N_WORD + 7) / 8, 256, 0, stream>>>(W2b, wl_word, bl_word, oWl, N_WORD);
}
